// temporal_vlstm_66374424593214
// MI455X (gfx1250) — matrix operand order repaired and checked by reading, not compiled, not run on hardware
//
#include <hip/hip_runtime.h>

// Problem constants (from reference)
#define BB 256
#define VV 16
#define TT 64
#define PP 24
#define FF 8
#define HH 256
#define G4 (4 * HH)
#define OUTF 4

typedef __attribute__((ext_vector_type(16))) __bf16 v16bf;
typedef __attribute__((ext_vector_type(8)))  float  v8f;
typedef unsigned int u32x4 __attribute__((ext_vector_type(4)));
typedef int          i32x4 __attribute__((ext_vector_type(4)));
typedef int          i32x8 __attribute__((ext_vector_type(8)));

union BfTile {
    v16bf v;
    unsigned short s[16];
    uint4 q[2];
};

__device__ __forceinline__ unsigned short f2bf(float f) {
    unsigned int u = __float_as_uint(f);
    u += 0x7fffu + ((u >> 16) & 1u);   // round-to-nearest-even
    return (unsigned short)(u >> 16);
}
__device__ __forceinline__ float bf2f(unsigned short s) {
    return __uint_as_float(((unsigned int)s) << 16);
}
__device__ __forceinline__ float sigmoidf_(float x) {
    return 1.0f / (1.0f + __expf(-x));
}

// A operand (16-bit A 16x32, ISA 7.12.2): lane&15 = row M, lane>>4 selects the
// K-half. p points at element K = k0 + 8*(lane>>4) of a contiguous K row:
// s[0..7] <- p[0..7], s[8..15] <- p[16..23].
__device__ __forceinline__ void load_a(BfTile& t, const unsigned short* p) {
    t.q[0] = *reinterpret_cast<const uint4*>(p);
    t.q[1] = *reinterpret_cast<const uint4*>(p + 16);
}

// ---------------- Tensor Data Mover helper ----------------
// Builds a D# per ISA 8.3/8.4 and issues TENSOR_LOAD_TO_LDS.
//   global tile:  addr + dsz*(x + y*stride0 + z*stride1), x<tile0, y<tile1,
//                 z<tile2 (tile1/tile2 == 0 -> unused)
//   LDS: packed x-fastest at ldsOff. data_size = 2 bytes (bf16).
__device__ __forceinline__ void tdm_load_bf16(
    unsigned long long gaddr, unsigned int ldsOff,
    unsigned int tile0, unsigned int tile1, unsigned int tile2,
    unsigned long long stride0, unsigned long long stride1,
    unsigned int td0, unsigned int td1, unsigned int td2) {
    u32x4 g0;
    i32x8 g1;
    i32x4 g2, g3;
    g0[0] = 1u;                                   // count=1, user descriptor
    g0[1] = ldsOff;                               // lds_addr (bytes)
    g0[2] = (unsigned int)gaddr;                  // global_addr[31:0]
    g0[3] = ((unsigned int)(gaddr >> 32) & 0x01FFFFFFu) | (2u << 30); // type=2
    g1[0] = 0x00010000;                           // data_size=1 (2B), no mcast
    g1[1] = (int)((td0 & 0xFFFFu) << 16);         // tensor_dim0[15:0] @ bit48
    g1[2] = (int)((td0 >> 16) | ((td1 & 0xFFFFu) << 16));
    g1[3] = (int)((td1 >> 16) | (tile0 << 16));   // tile_dim0 @ bit112
    g1[4] = (int)(tile1 | (tile2 << 16));         // tile_dim1, tile_dim2
    g1[5] = (int)(unsigned int)stride0;           // tensor_dim0_stride[31:0]
    g1[6] = (int)(((unsigned int)(stride0 >> 32) & 0xFFFFu) |
                  (((unsigned int)stride1 & 0xFFFFu) << 16));
    g1[7] = (int)(unsigned int)(stride1 >> 16);   // tensor_dim1_stride[47:16]
    g2[0] = (int)td2;                             // tensor_dim2
    g2[1] = 0; g2[2] = 0; g2[3] = 0;
    g3[0] = 0; g3[1] = 0; g3[2] = 0; g3[3] = 0;
#if __has_include(<hip/amd_detail/amd_gfx1250_TDM.h>)
    i32x8 gz = (i32x8){0, 0, 0, 0, 0, 0, 0, 0};
    __builtin_amdgcn_tensor_load_to_lds(g0, g1, g2, g3, gz, 0);
#else
    __builtin_amdgcn_tensor_load_to_lds(g0, g1, g2, g3, 0);
#endif
}

// ---------------- one-time prep kernels ----------------

// out[v][k][j] = bf16(whh[v][j][k]);  j in [0,4H), k in [0,H)
__global__ void cvt_whh_t_kernel(const float* __restrict__ whh,
                                 unsigned short* __restrict__ out) {
    long i = (long)blockIdx.x * blockDim.x + threadIdx.x;
    const long n = (long)VV * HH * G4;
    if (i >= n) return;
    int j = (int)(i % G4);
    long r = i / G4;
    int k = (int)(r % HH);
    int v = (int)(r / HH);
    out[i] = f2bf(whh[((long)v * G4 + j) * HH + k]);
}

// out[v][k][j] = bf16(watt[v][j][k]);  j in [0,H), k in [0,2H)
__global__ void cvt_watt_t_kernel(const float* __restrict__ watt,
                                  unsigned short* __restrict__ out) {
    long i = (long)blockIdx.x * blockDim.x + threadIdx.x;
    const long n = (long)VV * 2 * HH * HH;
    if (i >= n) return;
    int j = (int)(i % HH);
    long r = i / HH;
    int k = (int)(r % (2 * HH));
    int v = (int)(r / (2 * HH));
    out[i] = f2bf(watt[((long)v * HH + j) * (2 * HH) + k]);
}

__global__ void init_state_kernel(float* __restrict__ h, float* __restrict__ c,
                                  unsigned short* __restrict__ hbf, int n) {
    int i = blockIdx.x * blockDim.x + threadIdx.x;
    if (i < n) { h[i] = 0.0f; c[i] = 0.0f; hbf[i] = 0; }
}

__global__ void init_prev_kernel(const float* __restrict__ seq,
                                 float* __restrict__ prev) {
    int i = blockIdx.x * blockDim.x + threadIdx.x;   // over B*V*F
    if (i >= BB * VV * FF) return;
    int f = i % FF;
    int bv = i / FF;
    prev[i] = seq[((long)bv * TT + (TT - 1)) * FF + f];  // sequence[:,:,-1,:]
}

// ---------------- fused LSTM step (TDM + WMMA) ----------------
// MODE 0: encoder (x from sequence[:,:,t,:], A = h_bf, writes enc_bf)
// MODE 1: decoder (x = prev*(mx-mn)+mn,      A = hatt_bf)
// grid = (V, B/128, H/16), block = 256 (8 waves). Wave 0 TDM-stages the four
// gate weight tiles for this n-tile (16 cols x 4 gates x 256 K) into LDS as
// [k][gate][16]; each wave then owns a 16(b)x16(h) tile, keeps all four gate
// accumulators, and fuses the recurrence in-register.
template <int MODE>
__global__ __launch_bounds__(256) void lstm_step_kernel(
    const unsigned short* __restrict__ hin_bf,   // A operand source
    const unsigned short* __restrict__ whh_t,    // [v][k][4H] bf16
    const float* __restrict__ wih,               // (V,4H,F) f32
    const float* __restrict__ bih,
    const float* __restrict__ bhh,
    const float* __restrict__ seq,               // MODE 0
    const float* __restrict__ prev,              // MODE 1
    const float* __restrict__ maxv,              // MODE 1 (B,F)
    const float* __restrict__ minv,              // MODE 1 (B,F)
    float* __restrict__ h_f32,
    unsigned short* __restrict__ h_bf_out,
    float* __restrict__ c_f32,
    unsigned short* __restrict__ enc_bf,         // MODE 0
    int t) {
    __shared__ unsigned short ldsB[HH * 64];     // [k][g*16+col], 32 KB
    const int v    = blockIdx.x;
    const int n0   = blockIdx.z * 16;
    const int wave = threadIdx.x >> 5;
    const int lane = threadIdx.x & 31;
    const int m0   = blockIdx.y * 128 + wave * 16;
    const int l15  = lane & 15;
    const int hiK  = (lane >> 4) * 8;
    const int row  = m0 + l15;        // batch row for A operand
    const int n    = n0 + l15;        // output column (per-lane N)

    // Stage weights: x = 16 cols (contig), y = 4 gates (stride H),
    // z = 256 K rows (stride 4H). LDS gets [k][gate][16] packed.
    if (threadIdx.x < 32) {
        unsigned long long gaddr =
            (unsigned long long)(const void*)whh_t +
            2ull * ((unsigned long long)v * HH * G4 + n0);
        unsigned int ldsOff = (unsigned int)(unsigned long long)(const void*)ldsB;
        tdm_load_bf16(gaddr, ldsOff,
                      /*tile*/ 16u, 4u, (unsigned)HH,
                      /*strides*/ (unsigned long long)HH, (unsigned long long)G4,
                      /*tensor dims*/ (unsigned)G4, (unsigned)(G4 * 4), (unsigned)HH);
        __builtin_amdgcn_s_wait_tensorcnt(0);
    }
    __syncthreads();

    v8f acc[4];
#pragma unroll
    for (int g = 0; g < 4; ++g) acc[g] = (v8f){0, 0, 0, 0, 0, 0, 0, 0};

    const unsigned short* aBase = hin_bf + (size_t)(row * VV + v) * HH;
    for (int k0 = 0; k0 < HH; k0 += 32) {
        BfTile a;
        load_a(a, aBase + (k0 + hiK));
        // B operand: lane&15 = column N; 16 K values (table order) from LDS.
#pragma unroll
        for (int g = 0; g < 4; ++g) {
            BfTile b;
#pragma unroll
            for (int s = 0; s < 16; ++s)
                b.s[s] = ldsB[(k0 + hiK + (s < 8 ? s : s + 8)) * 64 + g * 16 + l15];
            acc[g] = __builtin_amdgcn_wmma_f32_16x16x32_bf16(
                false, a.v, false, b.v, (short)0, acc[g], false, false);
        }
    }

    // Per-lane (fixed gate column n): input-weights row + bias sums.
    float wihv[4][FF], bsum[4];
#pragma unroll
    for (int g = 0; g < 4; ++g) {
        const float* wr = wih + ((size_t)v * G4 + g * HH + n) * FF;
#pragma unroll
        for (int f = 0; f < FF; ++f) wihv[g][f] = wr[f];
        bsum[g] = bih[v * G4 + g * HH + n] + bhh[v * G4 + g * HH + n];
    }

#pragma unroll
    for (int r = 0; r < 8; ++r) {
        const int m = m0 + (lane >> 4) * 8 + r;  // C/D layout: lanes16-31 -> M+8
        float xf[FF];
        if (MODE == 0) {
            const float* xs = seq + (((size_t)(m * VV + v)) * TT + t) * FF;
#pragma unroll
            for (int f = 0; f < FF; ++f) xf[f] = xs[f];
        } else {
            const float* pv = prev + (size_t)(m * VV + v) * FF;
#pragma unroll
            for (int f = 0; f < FF; ++f) {
                float mx = maxv[m * FF + f], mn = minv[m * FF + f];
                xf[f] = pv[f] * (mx - mn) + mn;
            }
        }
        float gv[4];
#pragma unroll
        for (int g = 0; g < 4; ++g) {
            float s = acc[g][r] + bsum[g];
#pragma unroll
            for (int f = 0; f < FF; ++f) s += xf[f] * wihv[g][f];
            gv[g] = s;
        }
        const float ig = sigmoidf_(gv[0]);
        const float fg = sigmoidf_(gv[1]);
        const float gg = tanhf(gv[2]);
        const float og = sigmoidf_(gv[3]);
        const size_t idx = (size_t)(m * VV + v) * HH + n;
        const float cn = fg * c_f32[idx] + ig * gg;
        const float hn = og * tanhf(cn);
        c_f32[idx] = cn;
        h_f32[idx] = hn;
        h_bf_out[idx] = f2bf(hn);
        if (MODE == 0)
            enc_bf[(((size_t)(m * VV + v)) * TT + t) * HH + n] = f2bf(hn);
    }
}

// ---------------- attention (score / softmax / context) ----------------
// one 256-thread workgroup per (b,v)
__global__ __launch_bounds__(256) void attention_kernel(
    const unsigned short* __restrict__ enc_bf,
    const float* __restrict__ h_f32,
    const float* __restrict__ mask,
    unsigned short* __restrict__ cv_bf) {
    __shared__ float hs[HH];
    __shared__ float red[256];
    __shared__ float sc[TT];
    const int tid = threadIdx.x;
    const int bv  = blockIdx.x;

    hs[tid] = h_f32[(size_t)bv * HH + tid];
    __syncthreads();

    const int t  = tid >> 2;
    const int ch = tid & 3;
    const unsigned short* e = enc_bf + ((size_t)bv * TT + t) * HH + ch * 64;
    float p = 0.0f;
#pragma unroll 8
    for (int j = 0; j < 64; ++j) p += bf2f(e[j]) * hs[ch * 64 + j];
    red[tid] = p;
    __syncthreads();

    if (ch == 0)
        sc[t] = (red[tid] + red[tid + 1] + red[tid + 2] + red[tid + 3]) *
                mask[(size_t)bv * TT + t];
    __syncthreads();

    if (tid == 0) {
        float mx = sc[0];
        for (int k = 1; k < TT; ++k) mx = fmaxf(mx, sc[k]);
        float s = 0.0f;
        for (int k = 0; k < TT; ++k) { float e2 = __expf(sc[k] - mx); sc[k] = e2; s += e2; }
        const float inv = 1.0f / s;
        for (int k = 0; k < TT; ++k) sc[k] *= inv;
    }
    __syncthreads();

    float a = 0.0f;
    const unsigned short* ec = enc_bf + (size_t)bv * TT * HH + tid;
#pragma unroll 8
    for (int tt = 0; tt < TT; ++tt) a += sc[tt] * bf2f(ec[(size_t)tt * HH]);
    cv_bf[(size_t)bv * HH + tid] = f2bf(a);
}

// ---------------- attention projection: h_att = tanh(W_att @ [cv; h] + b) ----
// grid = (V, B/128, H/16), block = 256 (8 waves); TDM-staged weight tile
// (16 cols x 512 K rows -> 16 KB LDS), K = 2H = 512.
__global__ __launch_bounds__(256) void att_gemm_kernel(
    const unsigned short* __restrict__ cv_bf,
    const unsigned short* __restrict__ h_bf,
    const unsigned short* __restrict__ watt_t,   // [v][k(0..2H)][H] bf16
    const float* __restrict__ batt,
    unsigned short* __restrict__ hatt_bf) {
    __shared__ unsigned short ldsB[2 * HH * 16]; // [k][16], 16 KB
    const int v    = blockIdx.x;
    const int n0   = blockIdx.z * 16;
    const int wave = threadIdx.x >> 5;
    const int lane = threadIdx.x & 31;
    const int m0   = blockIdx.y * 128 + wave * 16;
    const int l15  = lane & 15;
    const int hiK  = (lane >> 4) * 8;
    const int row  = m0 + l15;
    const int n    = n0 + l15;

    if (threadIdx.x < 32) {
        unsigned long long gaddr =
            (unsigned long long)(const void*)watt_t +
            2ull * ((unsigned long long)v * (2 * HH) * HH + n0);
        unsigned int ldsOff = (unsigned int)(unsigned long long)(const void*)ldsB;
        tdm_load_bf16(gaddr, ldsOff,
                      /*tile*/ 16u, (unsigned)(2 * HH), 0u,
                      /*strides*/ (unsigned long long)HH, (unsigned long long)HH,
                      /*tensor dims*/ (unsigned)HH, (unsigned)(2 * HH * HH), 1u);
        __builtin_amdgcn_s_wait_tensorcnt(0);
    }
    __syncthreads();

    v8f acc = (v8f){0, 0, 0, 0, 0, 0, 0, 0};
    for (int k0 = 0; k0 < 2 * HH; k0 += 32) {
        BfTile a, b;
        const int kb = k0 + hiK;
        const unsigned short* aB =
            (k0 < HH) ? (cv_bf + (size_t)(row * VV + v) * HH + kb)
                      : (h_bf + (size_t)(row * VV + v) * HH + (kb - HH));
        load_a(a, aB);
#pragma unroll
        for (int s = 0; s < 16; ++s)
            b.s[s] = ldsB[(kb + (s < 8 ? s : s + 8)) * 16 + l15];
        acc = __builtin_amdgcn_wmma_f32_16x16x32_bf16(
            false, a.v, false, b.v, (short)0, acc, false, false);
    }
    const float bb = batt[v * HH + n];
#pragma unroll
    for (int r = 0; r < 8; ++r) {
        const int m = m0 + (lane >> 4) * 8 + r;
        hatt_bf[(size_t)(m * VV + v) * HH + n] = f2bf(tanhf(acc[r] + bb));
    }
}

// ---------------- output projection: out = relu(h2 @ W_out^T + b_out) --------
__global__ void out_proj_kernel(const float* __restrict__ h_f32,
                                const float* __restrict__ wout,   // (F,H)
                                const float* __restrict__ bout,
                                float* __restrict__ prev,
                                float* __restrict__ dout, int p) {
    int i = blockIdx.x * blockDim.x + threadIdx.x;   // over B*V*F
    if (i >= BB * VV * FF) return;
    const int f  = i % FF;
    const int bv = i / FF;
    const float* hb = h_f32 + (size_t)bv * HH;
    const float* w  = wout + (size_t)f * HH;
    float s = bout[f];
#pragma unroll 8
    for (int k = 0; k < HH; ++k) s += hb[k] * w[k];
    s = fmaxf(s, 0.0f);
    prev[i] = s;                                       // feedback (full F)
    if (f < OUTF)
        dout[((size_t)bv * PP + p) * OUTF + f] = fminf(s, 1.0f);
}

// ---------------- host-side orchestration ----------------
extern "C" void kernel_launch(void* const* d_in, const int* in_sizes, int n_in,
                              void* d_out, int out_size, void* d_ws,
                              size_t ws_size, hipStream_t stream) {
    const float* seq  = (const float*)d_in[0];
    const float* mask = (const float*)d_in[4];
    const float* maxv = (const float*)d_in[5];
    const float* minv = (const float*)d_in[6];
    const float* wih  = (const float*)d_in[7];
    const float* whh  = (const float*)d_in[8];
    const float* bih  = (const float*)d_in[9];
    const float* bhh  = (const float*)d_in[10];
    const float* watt = (const float*)d_in[11];
    const float* batt = (const float*)d_in[12];
    const float* wout = (const float*)d_in[13];
    const float* bout = (const float*)d_in[14];
    float* outp = (float*)d_out;

    // Carve scratch out of d_ws (256B aligned regions; ~162 MB total).
    char* ws = (char*)d_ws;
    size_t off = 0;
    auto take = [&](size_t bytes) -> char* {
        char* pp = ws + off;
        off = (off + bytes + 255) & ~(size_t)255;
        return pp;
    };
    const size_t BVH = (size_t)BB * VV * HH;
    unsigned short* whh_t  = (unsigned short*)take((size_t)VV * HH * G4 * 2);
    unsigned short* watt_t = (unsigned short*)take((size_t)VV * 2 * HH * HH * 2);
    float* h_f32           = (float*)take(BVH * 4);
    float* c_f32           = (float*)take(BVH * 4);
    unsigned short* h_bf   = (unsigned short*)take(BVH * 2);
    unsigned short* hatt_bf= (unsigned short*)take(BVH * 2);
    unsigned short* cv_bf  = (unsigned short*)take(BVH * 2);
    float* prev            = (float*)take((size_t)BB * VV * FF * 4);
    unsigned short* enc_bf = (unsigned short*)take(BVH * TT * 2);
    (void)ws_size; (void)in_sizes; (void)n_in; (void)out_size;

    // One-time: bf16 + transpose weights, zero state, seed prev.
    {
        long n = (long)VV * HH * G4;
        cvt_whh_t_kernel<<<(unsigned)((n + 255) / 256), 256, 0, stream>>>(whh, whh_t);
        n = (long)VV * 2 * HH * HH;
        cvt_watt_t_kernel<<<(unsigned)((n + 255) / 256), 256, 0, stream>>>(watt, watt_t);
        init_state_kernel<<<(unsigned)((BVH + 255) / 256), 256, 0, stream>>>(
            h_f32, c_f32, h_bf, (int)BVH);
        init_prev_kernel<<<(BB * VV * FF + 255) / 256, 256, 0, stream>>>(seq, prev);
    }

    const dim3 gTile(VV, BB / 128, HH / 16);   // (16, 2, 16)

    // Encoder: 64 recurrent steps.
    for (int t = 0; t < TT; ++t) {
        lstm_step_kernel<0><<<gTile, 256, 0, stream>>>(
            h_bf, whh_t, wih, bih, bhh, seq, nullptr, nullptr, nullptr,
            h_f32, h_bf, c_f32, enc_bf, t);
    }

    // Decoder: 24 steps of attention + LSTM + output projection.
    for (int p = 0; p < PP; ++p) {
        attention_kernel<<<BB * VV, 256, 0, stream>>>(enc_bf, h_f32, mask, cv_bf);
        att_gemm_kernel<<<gTile, 256, 0, stream>>>(cv_bf, h_bf, watt_t, batt, hatt_bf);
        lstm_step_kernel<1><<<gTile, 256, 0, stream>>>(
            hatt_bf, whh_t, wih, bih, bhh, nullptr, prev, maxv, minv,
            h_f32, h_bf, c_f32, nullptr, 0);
        out_proj_kernel<<<(BB * VV * FF + 255) / 256, 256, 0, stream>>>(
            h_f32, wout, bout, prev, outp, p);
    }
}
